// GATv3_83958020702406
// MI455X (gfx1250) — matrix operand order repaired and checked by reading, not compiled, not run on hardware
//
#include <hip/hip_runtime.h>

typedef __attribute__((ext_vector_type(16))) _Float16 v16h;
typedef __attribute__((ext_vector_type(8)))  _Float16 v8h;
typedef __attribute__((ext_vector_type(8)))  float    v8f;

#define NND 2048
#define NE  65536
#define ETOT (NE + NND)   // 67584 = 64 * 1056 exactly
#define QC  17            // K+1 subspace columns

__device__ __forceinline__ float leakyf(float x, float s) { return x > 0.f ? x : s * x; }
__device__ __forceinline__ unsigned fkey(float x) {
    unsigned b = __float_as_uint(x);
    return (b & 0x80000000u) ? ~b : (b | 0x80000000u);
}
__device__ __forceinline__ float funkey(unsigned k) {
    unsigned b = (k & 0x80000000u) ? (k & 0x7fffffffu) : ~k;
    return __uint_as_float(b);
}
__device__ __forceinline__ int srcOf(int e, const int* esrc) { return e < NE ? esrc[e] : e - NE; }
__device__ __forceinline__ int dstOf(int e, const int* edst) { return e < NE ? edst[e] : e - NE; }

// CDNA5 async global->LDS copy, 16 bytes per lane. Tracked by ASYNCcnt.
// LDS dest address = low 32 bits of the generic pointer (LDS aperture offset).
__device__ __forceinline__ void async_copy16(const _Float16* g, _Float16* l) {
    unsigned loff = (unsigned)(uintptr_t)l;
    asm volatile("global_load_async_to_lds_b128 %0, %1, off"
                 :: "v"(loff), "v"(g) : "memory");
}
// Wait until at most 2 async ops outstanding (drain older stage, keep newest in flight).
__device__ __forceinline__ void wait_async2() {
    asm volatile("s_wait_asynccnt 0x2" ::: "memory");
}
__device__ __forceinline__ void wait_async0() {
    asm volatile("s_wait_asynccnt 0x0" ::: "memory");
}

// ---------------- elementwise helpers ----------------
__global__ void k_fill_f32(float* p, float v, int n) {
    int i = blockIdx.x * blockDim.x + threadIdx.x; if (i < n) p[i] = v;
}
__global__ void k_fill_u32(unsigned* p, unsigned v, int n) {
    int i = blockIdx.x * blockDim.x + threadIdx.x; if (i < n) p[i] = v;
}
__global__ void k_copy_f32(const float* a, float* b, int n) {
    int i = blockIdx.x * blockDim.x + threadIdx.x; if (i < n) b[i] = a[i];
}
__global__ void k_cvt(const float* a, _Float16* b, int n) {
    int i = blockIdx.x * blockDim.x + threadIdx.x; if (i < n) b[i] = (_Float16)a[i];
}
// [R,C] f32 -> [C,R] f16
__global__ void k_transT(const float* __restrict__ a, _Float16* __restrict__ b, int R, int C) {
    int i = blockIdx.x * blockDim.x + threadIdx.x;
    if (i < R * C) { int r = i / C, c = i - r * C; b[(size_t)c * R + r] = (_Float16)a[i]; }
}
__global__ void k_relu_cvt(const float* a, _Float16* b, int n) {
    int i = blockIdx.x * blockDim.x + threadIdx.x; if (i < n) b[i] = (_Float16)fmaxf(a[i], 0.f);
}
// c[j] = b1[j] + sum_i b0[i]*W1[i,j]
__global__ void k_colcomb(const float* b0, const float* W1, const float* b1, float* c, int dIn, int dOut) {
    int j = blockIdx.x * blockDim.x + threadIdx.x;
    if (j < dOut) {
        float s = b1[j];
        for (int i = 0; i < dIn; ++i) s += b0[i] * W1[(size_t)i * dOut + j];
        c[j] = s;
    }
}
// emL[f] = c_em[f] + colsum_{k<32} WcEm[k,f]    (self-loop attrs = ones(32))
__global__ void k_emloop(const float* WcEm, const float* c_em, float* emL, int dOut) {
    int f = blockIdx.x * blockDim.x + threadIdx.x;
    if (f < dOut) {
        float s = c_em[f];
        for (int k = 0; k < 32; ++k) s += WcEm[(size_t)k * dOut + f];
        emL[f] = s;
    }
}
// WcEm [32,dOut] f32 -> src/dst halves transposed f16 [dOut,32], K padded 16->32 with zeros
__global__ void k_emsplit(const float* WcEm, _Float16* WsT, _Float16* WdT, int dOut) {
    int i = blockIdx.x * blockDim.x + threadIdx.x;
    if (i < dOut * 32) {
        int n = i >> 5, k = i & 31;
        float s = (k < 16) ? WcEm[(size_t)k * dOut + n] : 0.f;
        float d = (k < 16) ? WcEm[(size_t)(k + 16) * dOut + n] : 0.f;
        WsT[i] = (_Float16)s; WdT[i] = (_Float16)d;
    }
}

// ---------------- eigen-feature approximation ----------------
__global__ void k_qinit(float* Z) {
    int i = blockIdx.x * blockDim.x + threadIdx.x;
    if (i < NND * QC) {
        unsigned h = (unsigned)i * 2654435761u;
        h ^= h >> 13; h *= 0x85ebca6bu; h ^= h >> 16;
        Z[i] = (float)(h & 0xFFFFu) / 65536.f - 0.5f;
    }
}
__global__ void k_spmv(const int* esrc, const int* edst, const float* Q, float* Z) {
    int i = blockIdx.x * blockDim.x + threadIdx.x;
    if (i < NE * QC) {
        int e = i / QC, j = i - e * QC;
        atomicAdd(&Z[(size_t)esrc[e] * QC + j], Q[(size_t)edst[e] * QC + j]);
    }
}
__global__ __launch_bounds__(256) void k_mgs(float* __restrict__ Z, float* __restrict__ Q) {
    __shared__ float red[256];
    const int tid = threadIdx.x;
    for (int j = 0; j < QC; ++j) {
        float s = 0.f;
        for (int i = tid; i < NND; i += 256) { float v = Z[i * QC + j]; s += v * v; }
        red[tid] = s; __syncthreads();
        for (int st = 128; st > 0; st >>= 1) { if (tid < st) red[tid] += red[tid + st]; __syncthreads(); }
        float inv = rsqrtf(red[0] + 1e-20f); __syncthreads();
        for (int i = tid; i < NND; i += 256) Q[i * QC + j] = Z[i * QC + j] * inv;
        __syncthreads();
        for (int k = j + 1; k < QC; ++k) {
            s = 0.f;
            for (int i = tid; i < NND; i += 256) s += Q[i * QC + j] * Z[i * QC + k];
            red[tid] = s; __syncthreads();
            for (int st = 128; st > 0; st >>= 1) { if (tid < st) red[tid] += red[tid + st]; __syncthreads(); }
            float d = red[0]; __syncthreads();
            for (int i = tid; i < NND; i += 256) Z[i * QC + k] -= d * Q[i * QC + j];
            __syncthreads();
        }
    }
}
// top16pad[n][k] = Q[n][k+1] for k<16, else 0   (padded K=32 for WMMA)
__global__ void k_top(const float* Q, _Float16* top16) {
    int i = blockIdx.x * blockDim.x + threadIdx.x;
    if (i < NND * 32) {
        int n = i >> 5, k = i & 31;
        top16[i] = (k < 16) ? (_Float16)Q[n * QC + k + 1] : (_Float16)0.f;
    }
}

// ---------------- generic WMMA GEMM: C[M,Nc] = A16[M,K] @ BT16[Nc,K] (+bias) ----------------
// Triple-buffered GLOBAL_LOAD_ASYNC_TO_LDS_B128 pipeline:
//   issue(s+1) -> s_wait_asynccnt 2 (drain stage s only) -> barrier -> WMMA(stage s)
// Depth-3 buffers make issue-before-barrier WAR-safe with one barrier per stage.
// Loads are unguarded: callers guarantee rows up to the 64-row tile boundary are readable
// (workspace buffers are oversized); rows >= M only feed discarded outputs.
__global__ __launch_bounds__(256) void k_gemm(const _Float16* __restrict__ A,
                                              const _Float16* __restrict__ BT,
                                              const float* __restrict__ bias,
                                              float* __restrict__ C,
                                              int M, int Nc, int Kd) {
    __shared__ __attribute__((aligned(32))) _Float16 As[3][64 * 32];
    __shared__ __attribute__((aligned(32))) _Float16 Bs[3][64 * 32];
    const int tid = threadIdx.x;
    const int m0 = blockIdx.x * 64, n0 = blockIdx.y * 64;
    const int w = tid >> 5, lane = tid & 31, lm = lane & 15, lg = lane >> 4;
    const int mt = w & 3, ntA = (w >> 2), ntB = ntA + 2;
    const int lr = tid >> 2, lc = (tid & 3) * 8;
    const _Float16* gA = A + (size_t)(m0 + lr) * Kd + lc;
    const _Float16* gB = BT + (size_t)(n0 + lr) * Kd + lc;
    v8f acc0 = {}; v8f acc1 = {};
    const int nS = Kd >> 5;
    const int ldsIdx = lr * 32 + lc;
    async_copy16(gA, &As[0][ldsIdx]);
    async_copy16(gB, &Bs[0][ldsIdx]);
    int cur = 0, nxt = 1;
    for (int s = 0; s < nS; ++s) {
        if (s + 1 < nS) {
            const int k1 = (s + 1) << 5;
            async_copy16(gA + k1, &As[nxt][ldsIdx]);
            async_copy16(gB + k1, &Bs[nxt][ldsIdx]);
            wait_async2();        // older stage done; newest pair stays in flight
        } else {
            wait_async0();
        }
        __syncthreads();
        const _Float16* Asb = As[cur];
        const _Float16* Bsb = Bs[cur];
        v16h a;
        {
            a = *(const v16h*)(Asb + (mt * 16 + lm) * 32 + 16 * lg);
        }
        v16h b0 = *(const v16h*)(Bsb + (ntA * 16 + lm) * 32 + 16 * lg);
        v16h b1 = *(const v16h*)(Bsb + (ntB * 16 + lm) * 32 + 16 * lg);
        acc0 = __builtin_amdgcn_wmma_f32_16x16x32_f16(false, a, false, b0, (short)0, acc0, false, false);
        acc1 = __builtin_amdgcn_wmma_f32_16x16x32_f16(false, a, false, b1, (short)0, acc1, false, false);
        cur = nxt; nxt = (nxt == 2) ? 0 : nxt + 1;
        __builtin_amdgcn_sched_barrier(0);   // keep next stage's wait from hoisting above WMMAs
    }
    const int gmb = m0 + mt * 16 + 8 * lg;
    const int gn0 = n0 + ntA * 16 + lm, gn1 = n0 + ntB * 16 + lm;
    const float bb0 = bias ? bias[gn0] : 0.f;
    const float bb1 = bias ? bias[gn1] : 0.f;
    #pragma unroll
    for (int r = 0; r < 8; ++r) {
        int gm = gmb + r;
        if (gm < M) {
            C[(size_t)gm * Nc + gn0] = acc0[r] + bb0;
            C[(size_t)gm * Nc + gn1] = acc1[r] + bb1;
        }
    }
}

// ---------------- fused per-edge dual-GEMM + proj -> logit ----------------
// t2 = leaky02(U[dst]+V[src]+c_om), s2 = leaky02(ES[src]+ED[dst]+c_em | emL)
// z  = alpha*(t2@W2om + b2om) + beta*(s2@W2em + b2em)
// logit = leaky01(z) . projW + projb
__global__ __launch_bounds__(256) void k_edge_logit(
    const int* __restrict__ esrc, const int* __restrict__ edst,
    const float* __restrict__ U, const float* __restrict__ V,
    const float* __restrict__ ES, const float* __restrict__ ED,
    const float* __restrict__ c_om, const float* __restrict__ c_em, const float* __restrict__ emL,
    const _Float16* __restrict__ W2omT, const _Float16* __restrict__ W2emT,
    const float* __restrict__ b2om, const float* __restrict__ b2em,
    const float* __restrict__ alpha_p, const float* __restrict__ beta_p,
    const float* __restrict__ projW, const float* __restrict__ projb_p,
    float* __restrict__ logit, int dOut) {
    extern __shared__ __attribute__((aligned(32))) char smem[];
    _Float16* t2p = (_Float16*)smem;
    _Float16* s2p = t2p + 64 * dOut;
    float* lred = (float*)(s2p + 64 * dOut);
    const int tid = threadIdx.x;
    const int base = blockIdx.x * 64;   // grid is exactly ETOT/64 blocks
    const float alpha = alpha_p[0], beta = beta_p[0];
    if (tid < 64) lred[tid] = 0.f;
    for (int idx = tid; idx < 64 * dOut; idx += 256) {
        int r = idx / dOut, f = idx - r * dOut;
        int e = base + r;
        int s, d;
        if (e < NE) { s = esrc[e]; d = edst[e]; } else { s = d = e - NE; }
        float t1 = U[(size_t)d * dOut + f] + V[(size_t)s * dOut + f] + c_om[f];
        float s1 = (e < NE) ? (ES[(size_t)s * dOut + f] + ED[(size_t)d * dOut + f] + c_em[f]) : emL[f];
        t2p[idx] = (_Float16)leakyf(t1, 0.02f);
        s2p[idx] = (_Float16)leakyf(s1, 0.02f);
    }
    __syncthreads();
    const int w = tid >> 5, lane = tid & 31, lm = lane & 15, lg = lane >> 4;
    const int mt = w & 3;
    const int am = mt * 16 + lm;
    float part[8];
    #pragma unroll
    for (int r = 0; r < 8; ++r) part[r] = 0.f;
    const int nNT = dOut >> 4;
    for (int nt = (w >> 2); nt < nNT; nt += 2) {
        v8f aco = {}; v8f ace = {};
        const int n = nt * 16 + lm;
        const _Float16* rowO = W2omT + (size_t)n * dOut + 16 * lg;
        const _Float16* rowE = W2emT + (size_t)n * dOut + 16 * lg;
        for (int k0 = 0; k0 < dOut; k0 += 32) {
            if (k0 + 32 < dOut) {            // prefetch next K block (global_prefetch_b8)
                __builtin_prefetch(rowO + k0 + 32, 0, 0);
                __builtin_prefetch(rowE + k0 + 32, 0, 0);
            }
            v16h ao, ae;
            {
                ao = *(const v16h*)(t2p + am * dOut + k0 + 16 * lg);
                ae = *(const v16h*)(s2p + am * dOut + k0 + 16 * lg);
            }
            v16h bo = *(const v16h*)(rowO + k0);
            v16h be = *(const v16h*)(rowE + k0);
            aco = __builtin_amdgcn_wmma_f32_16x16x32_f16(false, ao, false, bo, (short)0, aco, false, false);
            ace = __builtin_amdgcn_wmma_f32_16x16x32_f16(false, ae, false, be, (short)0, ace, false, false);
        }
        const float bbo = b2om[n], bbe = b2em[n], pw = projW[n];
        #pragma unroll
        for (int r = 0; r < 8; ++r) {
            float z = alpha * (aco[r] + bbo) + beta * (ace[r] + bbe);
            part[r] += leakyf(z, 0.01f) * pw;
        }
    }
    #pragma unroll
    for (int r = 0; r < 8; ++r) atomicAdd(&lred[mt * 16 + r + 8 * lg], part[r]);
    __syncthreads();
    if (tid < 64) logit[base + tid] = lred[tid] + projb_p[0];
}

// ---------------- segment softmax + aggregation ----------------
__global__ void k_segmax(const int* esrc, const int* edst, const float* logit, unsigned* mkey) {
    int i = blockIdx.x * blockDim.x + threadIdx.x;
    if (i < ETOT) atomicMax(&mkey[dstOf(i, edst)], fkey(logit[i]));
}
__global__ void k_expsum(const int* esrc, const int* edst, const float* logit,
                         const unsigned* mkey, float* expw, float* ssum) {
    int i = blockIdx.x * blockDim.x + threadIdx.x;
    if (i < ETOT) {
        int d = dstOf(i, edst);
        float wv = __expf(logit[i] - funkey(mkey[d]));
        expw[i] = wv;
        atomicAdd(&ssum[d], wv);
    }
}
__global__ void k_gamma(const int* esrc, const int* edst, float* expw, const float* ssum) {
    int i = blockIdx.x * blockDim.x + threadIdx.x;
    if (i < ETOT) expw[i] = expw[i] / (ssum[dstOf(i, edst)] + 1e-16f);
}
__global__ void k_agg(const int* esrc, const int* edst, const float* expw,
                      const float* __restrict__ OUTn, float* __restrict__ out, int dOut) {
    int i = blockIdx.x * blockDim.x + threadIdx.x;
    if (i < ETOT * dOut) {
        int e = i / dOut, f = i - e * dOut;
        int s = srcOf(e, esrc), d = dstOf(e, edst);
        atomicAdd(&out[(size_t)d * dOut + f], expw[e] * OUTn[(size_t)s * dOut + f]);
    }
}

// ---------------- host orchestration ----------------
extern "C" void kernel_launch(void* const* d_in, const int* in_sizes, int n_in,
                              void* d_out, int out_size, void* d_ws, size_t ws_size,
                              hipStream_t stream) {
    (void)in_sizes; (void)n_in; (void)out_size; (void)ws_size;
    const float* x = (const float*)d_in[0];
    const int* eidx = (const int*)d_in[1];
    const int* esrc = eidx;
    const int* edst = eidx + NE;

    struct Lin { const float *W, *b; };
    struct LayerP { Lin om[3], em[3], proj, out; const float *alpha, *beta; };
    LayerP Lp[2];
    int ip = 2;
    for (int l = 0; l < 2; ++l) {
        for (int j = 0; j < 3; ++j) { Lp[l].om[j].W = (const float*)d_in[ip++]; Lp[l].om[j].b = (const float*)d_in[ip++]; }
        for (int j = 0; j < 3; ++j) { Lp[l].em[j].W = (const float*)d_in[ip++]; Lp[l].em[j].b = (const float*)d_in[ip++]; }
        Lp[l].proj.W = (const float*)d_in[ip++]; Lp[l].proj.b = (const float*)d_in[ip++];
        Lp[l].out.W  = (const float*)d_in[ip++]; Lp[l].out.b  = (const float*)d_in[ip++];
        Lp[l].alpha  = (const float*)d_in[ip++]; Lp[l].beta   = (const float*)d_in[ip++];
    }

    char* base = (char*)d_ws; size_t off = 0;
    auto alloc = [&](size_t bytes) -> void* {
        void* p = base + off; off = (off + bytes + 255) & ~(size_t)255; return p;
    };
    float*     Q      = (float*)alloc((size_t)NND * QC * 4);
    float*     Zb     = (float*)alloc((size_t)NND * QC * 4);
    _Float16*  top16  = (_Float16*)alloc((size_t)NND * 32 * 2);
    _Float16*  x16    = (_Float16*)alloc((size_t)NND * 512 * 2);
    _Float16*  A16    = (_Float16*)alloc((size_t)512 * 512 * 2);
    _Float16*  W1T16  = (_Float16*)alloc((size_t)512 * 512 * 2);
    float*     Wtmp32 = (float*)alloc((size_t)512 * 512 * 4);
    float*     WcEm32 = (float*)alloc((size_t)64 * 512 * 4);   // 64 rows so tile over-reads stay in-buffer
    _Float16*  WomdT  = (_Float16*)alloc((size_t)512 * 512 * 2);
    _Float16*  WomsT  = (_Float16*)alloc((size_t)512 * 512 * 2);
    _Float16*  WemsT  = (_Float16*)alloc((size_t)512 * 32 * 2);
    _Float16*  WemdT  = (_Float16*)alloc((size_t)512 * 32 * 2);
    _Float16*  W2omT  = (_Float16*)alloc((size_t)512 * 512 * 2);
    _Float16*  W2emT  = (_Float16*)alloc((size_t)512 * 512 * 2);
    _Float16*  WoutT  = (_Float16*)alloc((size_t)512 * 512 * 2);
    float*     c_om   = (float*)alloc(512 * 4);
    float*     c_em   = (float*)alloc(512 * 4);
    float*     emL    = (float*)alloc(512 * 4);
    float*     U32    = (float*)alloc((size_t)NND * 512 * 4);
    float*     V32    = (float*)alloc((size_t)NND * 512 * 4);
    float*     ES32   = (float*)alloc((size_t)NND * 512 * 4);
    float*     ED32   = (float*)alloc((size_t)NND * 512 * 4);
    float*     OUTn32 = (float*)alloc((size_t)NND * 512 * 4);
    float*     agg    = (float*)alloc((size_t)NND * 512 * 4);
    float*     logit  = (float*)alloc((size_t)ETOT * 4);
    float*     expw   = (float*)alloc((size_t)ETOT * 4);
    unsigned*  mkey   = (unsigned*)alloc((size_t)NND * 4);
    float*     ssum   = (float*)alloc((size_t)NND * 4);

    auto cdiv = [](int a, int b) { return (a + b - 1) / b; };
    #define L1(k, n, ...) k<<<cdiv((n), 256), 256, 0, stream>>>(__VA_ARGS__)

    // --- eigen features (orthogonal subspace iteration on A+I, deterministic) ---
    L1(k_qinit, NND * QC, Zb);
    k_mgs<<<1, 256, 0, stream>>>(Zb, Q);
    for (int it = 0; it < 16; ++it) {
        L1(k_copy_f32, NND * QC, Q, Zb, NND * QC);
        L1(k_spmv, NE * QC, esrc, edst, Q, Zb);
        k_mgs<<<1, 256, 0, stream>>>(Zb, Q);
    }
    L1(k_top, NND * 32, Q, top16);
    L1(k_cvt, NND * 256, x, x16, NND * 256);

    hipFuncSetAttribute((const void*)k_edge_logit,
                        hipFuncAttributeMaxDynamicSharedMemorySize, 64 * 512 * 4 + 256);

    for (int l = 0; l < 2; ++l) {
        const int dIn = l ? 512 : 256, dOut = l ? 256 : 512;
        const LayerP& P = Lp[l];
        // ---- weight prep: fold om/em layers 1+2 into single linears ----
        L1(k_cvt, dIn * dOut, P.om[0].W, A16, dIn * dOut);
        L1(k_transT, dOut * dOut, P.om[1].W, W1T16, dOut, dOut);
        k_gemm<<<dim3(cdiv(dIn, 64), dOut / 64), 256, 0, stream>>>(A16, W1T16, nullptr, Wtmp32, dIn, dOut, dOut);
        L1(k_transT, dIn * dOut, Wtmp32, WomdT, dIn, dOut);
        L1(k_cvt, dIn * dOut, P.om[0].W + (size_t)dIn * dOut, A16, dIn * dOut);
        k_gemm<<<dim3(cdiv(dIn, 64), dOut / 64), 256, 0, stream>>>(A16, W1T16, nullptr, Wtmp32, dIn, dOut, dOut);
        L1(k_transT, dIn * dOut, Wtmp32, WomsT, dIn, dOut);
        L1(k_colcomb, dOut, P.om[0].b, P.om[1].W, P.om[1].b, c_om, dOut, dOut);
        L1(k_cvt, 32 * dOut, P.em[0].W, A16, 32 * dOut);
        L1(k_transT, dOut * dOut, P.em[1].W, W1T16, dOut, dOut);
        k_gemm<<<dim3(1, dOut / 64), 256, 0, stream>>>(A16, W1T16, nullptr, WcEm32, 32, dOut, dOut);
        L1(k_colcomb, dOut, P.em[0].b, P.em[1].W, P.em[1].b, c_em, dOut, dOut);
        L1(k_emloop, dOut, WcEm32, c_em, emL, dOut);
        L1(k_emsplit, dOut * 32, WcEm32, WemsT, WemdT, dOut);
        L1(k_transT, dOut * dOut, P.om[2].W, W2omT, dOut, dOut);
        L1(k_transT, dOut * dOut, P.em[2].W, W2emT, dOut, dOut);
        L1(k_transT, dIn * dOut, P.out.W, WoutT, dIn, dOut);
        // ---- per-node GEMMs ----
        dim3 gn(NND / 64, dOut / 64);
        k_gemm<<<gn, 256, 0, stream>>>(x16, WomdT, nullptr, U32, NND, dOut, dIn);
        k_gemm<<<gn, 256, 0, stream>>>(x16, WomsT, nullptr, V32, NND, dOut, dIn);
        k_gemm<<<gn, 256, 0, stream>>>(top16, WemsT, nullptr, ES32, NND, dOut, 32);
        k_gemm<<<gn, 256, 0, stream>>>(top16, WemdT, nullptr, ED32, NND, dOut, 32);
        k_gemm<<<gn, 256, 0, stream>>>(x16, WoutT, P.out.b, OUTn32, NND, dOut, dIn);
        // ---- fused per-edge attention logit ----
        size_t shm = (size_t)64 * dOut * 2 * 2 + 64 * 4;
        k_edge_logit<<<ETOT / 64, 256, shm, stream>>>(esrc, edst, U32, V32, ES32, ED32,
            c_om, c_em, emL, W2omT, W2emT, P.om[2].b, P.em[2].b, P.alpha, P.beta,
            P.proj.W, P.proj.b, logit, dOut);
        // ---- segment softmax over destination ----
        L1(k_fill_u32, NND, mkey, 0u, NND);
        L1(k_fill_f32, NND, ssum, 0.f, NND);
        L1(k_segmax, ETOT, esrc, edst, logit, mkey);
        L1(k_expsum, ETOT, esrc, edst, logit, mkey, expw, ssum);
        L1(k_gamma, ETOT, esrc, edst, expw, ssum);
        // ---- weighted aggregation ----
        float* aggp = l ? (float*)d_out : agg;
        L1(k_fill_f32, NND * dOut, aggp, 0.f, NND * dOut);
        L1(k_agg, ETOT * dOut, esrc, edst, expw, OUTn32, aggp, dOut);
        if (!l) L1(k_relu_cvt, NND * dOut, agg, x16, NND * dOut);
    }
    #undef L1
}
